// ScanningMLP_88467736363291
// MI455X (gfx1250) — matrix operand order repaired and checked by reading, not compiled, not run on hardware
//
#include <hip/hip_runtime.h>
#include <hip/hip_bf16.h>

// ---------------------------------------------------------------------------
// ScanningMLP on gfx1250: 3x bf16 WMMA GEMMs (f32 accum) + sequential scan.
// Register-blocked (64x64 per wave32, 4x4 WMMA tiles), no LDS, no barriers.
// ---------------------------------------------------------------------------

typedef __attribute__((ext_vector_type(16))) __bf16 v16bf;
typedef __attribute__((ext_vector_type(8)))  float  v8f;

#define BATCH_N 8
#define SEQ_T   2048
#define D_M     1024
#define E_M     2048   // D_MODEL * EXPAND
#define S_N     256
#define MT      (BATCH_N * SEQ_T)   // 16384 rows total

// ---- WMMA A-fragment mapping (16x32 bf16, ISA 7.12.2) ----------------------
// lane = m + 16*hi ; element e covers the VGPR0..7 K interleave.
// Inverse: given (k_in_tile, m) produce (lane, e).
__device__ __forceinline__ void a_dst_map(int kin, int m, int& lane, int& e) {
    int v4  = (kin >= 16) ? 4 : 0;   // VGPRs 4..7 hold K>=16
    int rem = kin & 15;
    int hi  = rem >> 3;              // lanes 16..31 hold the +8 K group
    int r2  = rem & 7;
    int vv  = v4 + (r2 >> 1);
    int h   = r2 & 1;
    e    = vv * 2 + h;
    lane = m + 16 * hi;
}

__device__ __forceinline__ float silu_f(float v) {
    return v / (1.0f + __expf(-v));
}

// ---- Pack f32 activation [Mtot, K] -> bf16 WMMA A-fragment tiles -----------
// dst tile layout: [mtile][ktile][lane 0..31][e 0..15]  (512 bf16 per tile)
__global__ void pack_A_kernel(const float* __restrict__ src,
                              __bf16* __restrict__ dst, int K, long total) {
    long idx = (long)blockIdx.x * blockDim.x + threadIdx.x;
    if (idx >= total) return;
    int  k  = (int)(idx % K);
    long mg = idx / K;
    long mtile = mg >> 4;
    int  m     = (int)(mg & 15);
    int  kt  = k >> 5, kin = k & 31;
    int  lane, e;
    a_dst_map(kin, m, lane, e);
    long Ktiles = K >> 5;
    dst[(mtile * Ktiles + kt) * 512 + lane * 16 + e] = (__bf16)src[idx];
}

// ---- Pack f32 weight [K, N] (row-major) -> bf16 WMMA B-fragment tiles ------
// B 32x16: lane = n + 16*((k_in>>3)&1), element e -> K = 8*hi + e (e<8), 16 + 8*hi + (e-8) (e>=8).
// dst tile layout: [ntile][ktile][lane][e]
__global__ void pack_B_kernel(const float* __restrict__ src,
                              __bf16* __restrict__ dst, int K, int N, long total) {
    long idx = (long)blockIdx.x * blockDim.x + threadIdx.x;
    if (idx >= total) return;
    int  n = (int)(idx % N);
    long k = idx / N;
    int  ntile = n >> 4, nl = n & 15;
    int  kt = (int)(k >> 5), kin = (int)(k & 31);
    int  hi = (kin >> 3) & 1;
    int  e  = (kin & 7) + 8 * (kin >> 4);
    int  lane = nl + 16 * hi;
    long Ktiles = K >> 5;
    dst[((long)ntile * Ktiles + kt) * 512 + lane * 16 + e] = (__bf16)src[idx];
}

// ---- Register-blocked bf16 WMMA GEMM:  C = act(A @ W + bias) ---------------
// Block: 128 threads = 4 wave32 splitting the N dimension.
// Each wave owns a 64(M) x 64(N) C tile = 4x4 WMMA accumulators.
// Per K-step: 4 A-frag + 4 B-frag loads (32B/lane each) -> 16 WMMAs.
template <bool SILU, bool PACK_OUT>
__global__ __launch_bounds__(128)
void gemm_bf16_wmma(const __bf16* __restrict__ Ap,
                    const __bf16* __restrict__ Bp,
                    const float*  __restrict__ bias,
                    float*        __restrict__ Cf32,   // used if !PACK_OUT
                    __bf16*       __restrict__ Cpk,    // used if PACK_OUT
                    int Ktiles, int Ntotal, int outKtiles) {
    const int tid    = threadIdx.x;
    const int wave   = tid >> 5;
    const int lane   = tid & 31;
    const int mtile0 = blockIdx.x * 4;                 // 4 M-tiles per wave
    const int ntile0 = (blockIdx.y * 4 + wave) * 4;    // 4 N-tiles per wave
    const int n      = lane & 15;   // column within 16x16 C tile
    const int hic    = lane >> 4;   // C rows 8..15 live in lanes 16..31

    // bias depends only on N -> broadcast into all 8 accumulator VGPRs
    v8f acc[4][4];
#pragma unroll
    for (int j = 0; j < 4; ++j) {
        float bval = bias[(ntile0 + j) * 16 + n];
#pragma unroll
        for (int i = 0; i < 4; ++i)
#pragma unroll
            for (int r = 0; r < 8; ++r) acc[i][j][r] = bval;
    }

    // per-fragment base pointers (lane offset folded in)
    const __bf16* Arow[4];
    const __bf16* Brow[4];
#pragma unroll
    for (int i = 0; i < 4; ++i)
        Arow[i] = Ap + ((size_t)(mtile0 + i) * Ktiles) * 512 + lane * 16;
#pragma unroll
    for (int j = 0; j < 4; ++j)
        Brow[j] = Bp + ((size_t)(ntile0 + j) * Ktiles) * 512 + lane * 16;

    for (int kt = 0; kt < Ktiles; ++kt) {
        const size_t koff = (size_t)kt * 512;
        v16bf a[4], b[4];
#pragma unroll
        for (int i = 0; i < 4; ++i) a[i] = *(const v16bf*)(Arow[i] + koff);
#pragma unroll
        for (int j = 0; j < 4; ++j) b[j] = *(const v16bf*)(Brow[j] + koff);
        if (kt + 1 < Ktiles) {   // warm next A tiles (global_prefetch_b8)
#pragma unroll
            for (int i = 0; i < 4; ++i)
                __builtin_prefetch(Arow[i] + koff + 512, 0, 1);
        }
#pragma unroll
        for (int i = 0; i < 4; ++i)
#pragma unroll
            for (int j = 0; j < 4; ++j)
                acc[i][j] = __builtin_amdgcn_wmma_f32_16x16x32_bf16(
                    /*neg_a=*/false, a[i], /*neg_b=*/false, b[j],
                    /*c_mod=*/(short)0, acc[i][j],
                    /*reuse_a=*/false, /*reuse_b=*/false);
    }

#pragma unroll
    for (int i = 0; i < 4; ++i) {
#pragma unroll
        for (int j = 0; j < 4; ++j) {
            if (!PACK_OUT) {
                // plain f32 row-major [Mtot, Ntotal]
                const int row0 = (mtile0 + i) * 16 + 8 * hic;
                const int col  = (ntile0 + j) * 16 + n;
#pragma unroll
                for (int r = 0; r < 8; ++r) {
                    float v = acc[i][j][r];
                    if (SILU) v = silu_f(v);
                    Cf32[(size_t)(row0 + r) * Ntotal + col] = v;
                }
            } else {
                // write directly into the NEXT gemm's A-fragment layout
#pragma unroll
                for (int r = 0; r < 8; ++r) {
                    float v = acc[i][j][r];
                    if (SILU) v = silu_f(v);
                    int Ml   = r + 8 * hic;              // row within tile
                    int kcol = (ntile0 + j) * 16 + n;    // K of next GEMM
                    int kt2  = kcol >> 5, kin2 = kcol & 31;
                    int lane2, e2;
                    a_dst_map(kin2, Ml, lane2, e2);
                    Cpk[((size_t)(mtile0 + i) * outKtiles + kt2) * 512 +
                        lane2 * 16 + e2] = (__bf16)v;
                }
            }
        }
    }
}

// ---- Sequential scan: state = silu(state + u_t), states -> packed bf16 A ---
__global__ void scan_kernel(const float* __restrict__ updates,    // [MT, S]
                            const float* __restrict__ init_state, // [1, S]
                            __bf16* __restrict__ statesP) {
    int tid = blockIdx.x * blockDim.x + threadIdx.x;   // 2048 = B*S threads
    int b = tid / S_N;
    int s = tid % S_N;
    float state = init_state[s];
    const int ktS = s >> 5, kin = s & 31;
    const int KtilesS = S_N >> 5;   // 8
    for (int t = 0; t < SEQ_T; ++t) {
        int row = b * SEQ_T + t;
        float v = state + updates[(size_t)row * S_N + s];
        state = silu_f(v);
        int mtile = row >> 4, m = row & 15;
        int lane2, e2;
        a_dst_map(kin, m, lane2, e2);
        statesP[((size_t)mtile * KtilesS + ktS) * 512 + lane2 * 16 + e2] =
            (__bf16)state;
    }
}

// ---------------------------------------------------------------------------
extern "C" void kernel_launch(void* const* d_in, const int* in_sizes, int n_in,
                              void* d_out, int out_size, void* d_ws, size_t ws_size,
                              hipStream_t stream) {
    const float* x          = (const float*)d_in[0];
    const float* W_in       = (const float*)d_in[1];
    const float* b_in       = (const float*)d_in[2];
    const float* W_state    = (const float*)d_in[3];
    const float* b_state    = (const float*)d_in[4];
    const float* W_out      = (const float*)d_in[5];
    const float* b_out      = (const float*)d_in[6];
    const float* init_state = (const float*)d_in[7];
    float*       out        = (float*)d_out;

    // ---- workspace carve-up ----
    char* ws = (char*)d_ws;
    size_t off = 0;
    __bf16* xP     = (__bf16*)(ws + off); off += (size_t)MT * D_M * 2;   // 32 MB
    __bf16* WinP   = (__bf16*)(ws + off); off += (size_t)D_M * E_M * 2;  //  4 MB
    __bf16* WstP   = (__bf16*)(ws + off); off += (size_t)E_M * S_N * 2;  //  1 MB
    __bf16* WoutP  = (__bf16*)(ws + off); off += (size_t)S_N * D_M * 2;  // .5 MB
    __bf16* xprojP = (__bf16*)(ws + off); off += (size_t)MT * E_M * 2;   // 64 MB
    float*  upd    = (float*) (ws + off); off += (size_t)MT * S_N * 4;   // 16 MB
    __bf16* stP    = (__bf16*)(ws + off); off += (size_t)MT * S_N * 2;   //  8 MB

    // ---- 1) f32 -> bf16 fragment packing ----
    {
        long t = (long)MT * D_M;
        pack_A_kernel<<<(t + 255) / 256, 256, 0, stream>>>(x, xP, D_M, t);
    }
    {
        long t = (long)D_M * E_M;
        pack_B_kernel<<<(t + 255) / 256, 256, 0, stream>>>(W_in, WinP, D_M, E_M, t);
    }
    {
        long t = (long)E_M * S_N;
        pack_B_kernel<<<(t + 255) / 256, 256, 0, stream>>>(W_state, WstP, E_M, S_N, t);
    }
    {
        long t = (long)S_N * D_M;
        pack_B_kernel<<<(t + 255) / 256, 256, 0, stream>>>(W_out, WoutP, S_N, D_M, t);
    }

    const int mblocks = MT / 64;   // 256 (64 M rows per block)

    // ---- 2) GEMM1: x_proj = silu(x @ W_in + b_in), output pre-packed ----
    {
        dim3 grid(mblocks, E_M / 256);   // (256, 8)
        gemm_bf16_wmma<true, true><<<grid, 128, 0, stream>>>(
            xP, WinP, b_in, nullptr, xprojP,
            /*Ktiles=*/D_M / 32, /*Ntotal=*/E_M, /*outKtiles=*/E_M / 32);
    }

    // ---- 3) GEMM2: updates = x_proj @ W_state + b_state (f32) ----
    {
        dim3 grid(mblocks, S_N / 256);   // (256, 1)
        gemm_bf16_wmma<false, false><<<grid, 128, 0, stream>>>(
            xprojP, WstP, b_state, upd, nullptr,
            /*Ktiles=*/E_M / 32, /*Ntotal=*/S_N, /*outKtiles=*/0);
    }

    // ---- 4) sequential scan -> states (packed bf16 A for GEMM3) ----
    scan_kernel<<<(BATCH_N * S_N) / 256, 256, 0, stream>>>(upd, init_state, stP);

    // ---- 5) GEMM3: out = states @ W_out + b_out (f32) ----
    {
        dim3 grid(mblocks, D_M / 256);   // (256, 4)
        gemm_bf16_wmma<false, false><<<grid, 128, 0, stream>>>(
            stP, WoutP, b_out, out, nullptr,
            /*Ktiles=*/S_N / 32, /*Ntotal=*/D_M, /*outKtiles=*/0);
    }
    (void)in_sizes; (void)n_in; (void)out_size; (void)ws_size; (void)off;
}
